// ContrastiveLoss_82892868813397
// MI455X (gfx1250) — compile-verified, matrix operand order checked by reading, not run on hardware
//
#include <hip/hip_runtime.h>

typedef __attribute__((ext_vector_type(16))) __bf16 v16bf;
typedef __attribute__((ext_vector_type(8)))  __bf16 v8bf;
typedef __attribute__((ext_vector_type(8)))  float  v8f;

#define N_ROWS 4096
#define DIM    1024
#define EPS_N  1e-8f
#define MARGIN 0.5f
#define BM     128   // block tile rows (img)
#define BN     256   // block tile cols (txt)
#define KS     32    // K step == WMMA K
#define LDA    40    // padded LDS row stride (ushorts): 32 + 8

static __device__ __forceinline__ unsigned short f32_to_bf16(float f) {
    unsigned int u = __float_as_uint(f);
    u += 0x7FFFu + ((u >> 16) & 1u);     // round-to-nearest-even
    return (unsigned short)(u >> 16);
}

static __device__ __forceinline__ float wave_reduce_add(float v) {
#pragma unroll
    for (int off = 16; off > 0; off >>= 1)
        v += __shfl_xor(v, off, 32);
    return v;
}

// gfx1250 async global->LDS copy, GVS mode (SGPR base + 32-bit voffset).
// Tracked by ASYNCcnt; LDS dest address per lane in a VGPR.
static __device__ __forceinline__ void async_ld_b128(unsigned lds_addr,
                                                     unsigned goffset,
                                                     const void* base) {
    asm volatile("global_load_async_to_lds_b128 %0, %1, %2"
                 :: "v"(lds_addr), "v"(goffset), "s"(base)
                 : "memory");
}

// ---------------------------------------------------------------------------
// Kernel 1: per-row norms, positive loss term, bf16 normalized rows.
// ---------------------------------------------------------------------------
__global__ __launch_bounds__(256)
void normalize_kernel(const float* __restrict__ img,
                      const float* __restrict__ txt,
                      unsigned short* __restrict__ img_n,
                      unsigned short* __restrict__ txt_n,
                      float* __restrict__ pos_partial) {
    const int row  = blockIdx.x;
    const int tid  = threadIdx.x;
    const int base = row * DIM + tid * 4;

    const float4 a = *(const float4*)(img + base);
    const float4 b = *(const float4*)(txt + base);

    float si = a.x*a.x + a.y*a.y + a.z*a.z + a.w*a.w;
    float st = b.x*b.x + b.y*b.y + b.z*b.z + b.w*b.w;
    float dt = a.x*b.x + a.y*b.y + a.z*b.z + a.w*b.w;

    si = wave_reduce_add(si);
    st = wave_reduce_add(st);
    dt = wave_reduce_add(dt);

    __shared__ float red[3][8];
    __shared__ float bc[3];
    const int wid = tid >> 5, lane = tid & 31;
    if (lane == 0) { red[0][wid] = si; red[1][wid] = st; red[2][wid] = dt; }
    __syncthreads();
    if (tid < 3) {
        float s = 0.f;
#pragma unroll
        for (int i = 0; i < 8; ++i) s += red[tid][i];
        bc[tid] = s;
    }
    __syncthreads();

    const float norm_i = fmaxf(sqrtf(bc[0]), EPS_N);
    const float norm_t = fmaxf(sqrtf(bc[1]), EPS_N);

    if (tid == 0) {
        const float pos = bc[2] / (norm_i * norm_t);
        const float d   = 1.0f - pos;
        pos_partial[row] = d * d;
    }

    const float ri = 1.0f / norm_i;
    const float rt = 1.0f / norm_t;

    unsigned int i0 = (unsigned)f32_to_bf16(a.x * ri) | ((unsigned)f32_to_bf16(a.y * ri) << 16);
    unsigned int i1 = (unsigned)f32_to_bf16(a.z * ri) | ((unsigned)f32_to_bf16(a.w * ri) << 16);
    unsigned int t0 = (unsigned)f32_to_bf16(b.x * rt) | ((unsigned)f32_to_bf16(b.y * rt) << 16);
    unsigned int t1 = (unsigned)f32_to_bf16(b.z * rt) | ((unsigned)f32_to_bf16(b.w * rt) << 16);

    *(uint2*)(img_n + base) = make_uint2(i0, i1);
    *(uint2*)(txt_n + base) = make_uint2(t0, t1);
}

// ---------------------------------------------------------------------------
// Kernel 2: bf16 WMMA GEMM fused with hinge^2 epilogue.
// 8 waves = 2(M) x 4(N); block tile 128x256; wave tile 64x64 (16 WMMA tiles).
// Double-buffered LDS fed by GLOBAL_LOAD_ASYNC_TO_LDS_B128 (ASYNCcnt),
// issued one stage ahead so the copy hides under the WMMA block.
// ---------------------------------------------------------------------------
__global__ __launch_bounds__(256)
void simloss_gemm_kernel(const unsigned short* __restrict__ A,   // img_n
                         const unsigned short* __restrict__ B,   // txt_n
                         float* __restrict__ neg_partial) {
    __shared__ unsigned short As[2][BM][LDA];   // 2 x 128 x 40 u16 = 20 KB
    __shared__ unsigned short Bs[2][BN][LDA];   // 2 x 256 x 40 u16 = 40 KB
    __shared__ float wsum[8];

    const int tid  = threadIdx.x;
    const int lane = tid & 31;
    const int wid  = tid >> 5;      // 0..7
    const int wm   = wid & 1;       // 0..1 -> 64-row half of BM
    const int wn   = wid >> 1;      // 0..3 -> 64-col slice of BN

    const int rowBase = blockIdx.y * BM;
    const int colBase = blockIdx.x * BN;

    const int r  = lane & 15;       // row within 16x16 tile (A) / col (B,C)
    const int kh = lane >> 4;       // K-half select per ISA 16-bit layout

    const unsigned strideA = BM * LDA * 2;   // bytes between As buffers
    const unsigned strideB = BN * LDA * 2;   // bytes between Bs buffers

    // Per-thread staging chunks: A = 512 b128 chunks (2/thread), B = 1024 (4/thread)
    unsigned gA[2], lA[2];           // global byte offset (stage 0), LDS addr (buf 0)
    unsigned gB[4], lB[4];
#pragma unroll
    for (int i = 0; i < 2; ++i) {
        const int c  = tid + i * 256;
        const int rr = c >> 2, kk = (c & 3) * 8;
        gA[i] = (unsigned)(((rowBase + rr) * DIM + kk) * 2);
        lA[i] = (unsigned)(unsigned long long)(&As[0][rr][kk]);
    }
#pragma unroll
    for (int i = 0; i < 4; ++i) {
        const int c  = tid + i * 256;
        const int rr = c >> 2, kk = (c & 3) * 8;
        gB[i] = (unsigned)(((colBase + rr) * DIM + kk) * 2);
        lB[i] = (unsigned)(unsigned long long)(&Bs[0][rr][kk]);
    }

    v8f acc[4][4];
#pragma unroll
    for (int mi = 0; mi < 4; ++mi)
#pragma unroll
        for (int ni = 0; ni < 4; ++ni)
#pragma unroll
            for (int v = 0; v < 8; ++v) acc[mi][ni][v] = 0.0f;

    // prologue: stage 0 -> buffer 0 (6 async b128 per lane)
#pragma unroll
    for (int i = 0; i < 2; ++i) async_ld_b128(lA[i], gA[i], A);
#pragma unroll
    for (int i = 0; i < 4; ++i) async_ld_b128(lB[i], gB[i], B);
#pragma unroll
    for (int i = 0; i < 2; ++i) gA[i] += KS * 2;
#pragma unroll
    for (int i = 0; i < 4; ++i) gB[i] += KS * 2;

    const int nsteps = DIM / KS;    // 32
    for (int s = 0; s < nsteps; ++s) {
        const int cur = s & 1;

        if (s + 1 < nsteps) {
            __syncthreads();        // readers of buffer nxt (step s-1) are done
            const unsigned nxt = (unsigned)(cur ^ 1);
#pragma unroll
            for (int i = 0; i < 2; ++i) async_ld_b128(lA[i] + nxt * strideA, gA[i], A);
#pragma unroll
            for (int i = 0; i < 4; ++i) async_ld_b128(lB[i] + nxt * strideB, gB[i], B);
#pragma unroll
            for (int i = 0; i < 2; ++i) gA[i] += KS * 2;
#pragma unroll
            for (int i = 0; i < 4; ++i) gB[i] += KS * 2;
            asm volatile("s_wait_asynccnt 0x6" ::: "memory");  // stage s landed
        } else {
            asm volatile("s_wait_asynccnt 0x0" ::: "memory");
        }
        __syncthreads();            // stage-s data from ALL waves visible

        // compute from LDS buffer `cur`
        v16bf bfrag[4];
#pragma unroll
        for (int ni = 0; ni < 4; ++ni) {
            const int rr = wn * 64 + ni * 16 + r;
            v8bf lo = *(const v8bf*)(&Bs[cur][rr][kh * 8]);
            v8bf hi = *(const v8bf*)(&Bs[cur][rr][16 + kh * 8]);
            bfrag[ni] = __builtin_shufflevector(lo, hi,
                0,1,2,3,4,5,6,7,8,9,10,11,12,13,14,15);
        }
#pragma unroll
        for (int mi = 0; mi < 4; ++mi) {
            const int rr = wm * 64 + mi * 16 + r;
            v8bf lo = *(const v8bf*)(&As[cur][rr][kh * 8]);
            v8bf hi = *(const v8bf*)(&As[cur][rr][16 + kh * 8]);
            v16bf afrag = __builtin_shufflevector(lo, hi,
                0,1,2,3,4,5,6,7,8,9,10,11,12,13,14,15);
#pragma unroll
            for (int ni = 0; ni < 4; ++ni)
                acc[mi][ni] = __builtin_amdgcn_wmma_f32_16x16x32_bf16(
                    false, afrag, false, bfrag[ni],
                    (short)0, acc[mi][ni], false, false);
        }
    }

    // Epilogue: sum relu(sim - margin)^2, diagonal zeroed.
    float local = 0.f;
#pragma unroll
    for (int mi = 0; mi < 4; ++mi) {
#pragma unroll
        for (int ni = 0; ni < 4; ++ni) {
#pragma unroll
            for (int v = 0; v < 8; ++v) {
                const float sv = acc[mi][ni][v];
                const int   gr = rowBase + wm * 64 + mi * 16 + v + 8 * kh;
                const int   gc = colBase + wn * 64 + ni * 16 + r;
                float t = fmaxf(sv - MARGIN, 0.0f);
                float c = t * t;
                local += (gr == gc) ? 0.0f : c;
            }
        }
    }
    local = wave_reduce_add(local);
    if (lane == 0) wsum[wid] = local;
    __syncthreads();
    if (tid == 0) {
        float s = 0.f;
#pragma unroll
        for (int i = 0; i < 8; ++i) s += wsum[i];
        neg_partial[blockIdx.y * gridDim.x + blockIdx.x] = s;
    }
}

// ---------------------------------------------------------------------------
// Kernel 3: deterministic final reduction.
// ---------------------------------------------------------------------------
__global__ __launch_bounds__(256)
void finalize_kernel(const float* __restrict__ pos_partial,
                     const float* __restrict__ neg_partial,
                     int n_neg,
                     float* __restrict__ out) {
    const int tid = threadIdx.x;
    float p = 0.f, n = 0.f;
    for (int i = tid; i < N_ROWS; i += 256) p += pos_partial[i];
    for (int i = tid; i < n_neg;  i += 256) n += neg_partial[i];
    p = wave_reduce_add(p);
    n = wave_reduce_add(n);

    __shared__ float sp[8], sn[8];
    const int wid = tid >> 5, lane = tid & 31;
    if (lane == 0) { sp[wid] = p; sn[wid] = n; }
    __syncthreads();
    if (tid == 0) {
        float P = 0.f, Ng = 0.f;
#pragma unroll
        for (int i = 0; i < 8; ++i) { P += sp[i]; Ng += sn[i]; }
        out[0] = P / (float)N_ROWS + Ng / ((float)N_ROWS * (float)N_ROWS);
    }
}

extern "C" void kernel_launch(void* const* d_in, const int* in_sizes, int n_in,
                              void* d_out, int out_size, void* d_ws, size_t ws_size,
                              hipStream_t stream) {
    const float* img = (const float*)d_in[0];
    const float* txt = (const float*)d_in[1];

    char* ws = (char*)d_ws;
    unsigned short* img_n = (unsigned short*)ws;                                   // 8 MB
    unsigned short* txt_n = (unsigned short*)(ws + (size_t)N_ROWS * DIM * 2);      // 8 MB
    float* pos_partial    = (float*)(ws + (size_t)N_ROWS * DIM * 4);               // 16 KB
    float* neg_partial    = pos_partial + N_ROWS;                                  // 2 KB

    normalize_kernel<<<N_ROWS, 256, 0, stream>>>(img, txt, img_n, txt_n, pos_partial);

    dim3 grid(N_ROWS / BN, N_ROWS / BM);   // 16 x 32 tiles
    simloss_gemm_kernel<<<grid, 256, 0, stream>>>(img_n, txt_n, neg_partial);

    finalize_kernel<<<1, 256, 0, stream>>>(pos_partial, neg_partial,
                                           (N_ROWS / BN) * (N_ROWS / BM),
                                           (float*)d_out);
}
